// StateSpaceModel_6021544149696
// MI455X (gfx1250) — compile-verified, matrix operand order checked by reading, not run on hardware
//
#include <hip/hip_runtime.h>
#include <hip/hip_bf16.h>
#include <math.h>

typedef __bf16 bf16;
typedef __attribute__((ext_vector_type(16))) __bf16 v16bf;
typedef __attribute__((ext_vector_type(8)))  float  v8f;

// Problem dims (fixed by the reference)
#define BB 512
#define TT 256
#define ZZ 128
#define XX 64
#define RR 1024

// ---------------------------------------------------------------------------
// Small-tile bf16 WMMA GEMM:  C[M,N] = epi(A[M,K] (lda) * W[N,K]^T + bias[N])
// Block = 128 threads (4 waves). Wave tile = 32x32 (2x2 wmma accumulators).
// Block tile = 64(M) x 64(N). Grid = (N/64, M/64). Used for N in {64,128}.
// EPI: 0 = store f32 (acc+bias); 1 = store bf16 relu(acc+bias)
// ---------------------------------------------------------------------------
template <int EPI>
__global__ __launch_bounds__(128)
void gemm_wmma(const bf16* __restrict__ A, int lda,
               const bf16* __restrict__ W,
               const float* __restrict__ bias,
               void* __restrict__ Cout, int ldc, int K)
{
    const int wave  = threadIdx.x >> 5;
    const int lane  = threadIdx.x & 31;
    const int tileM = blockIdx.y * 64 + (wave >> 1) * 32;
    const int tileN = blockIdx.x * 64 + (wave & 1) * 32;

    const int rA = lane & 15;         // row within 16-row fragment
    const int kO = (lane >> 4) * 16;  // K half selected by upper lane group

    const bf16* Ap = A + (long)(tileM + rA) * lda + kO;
    const bf16* Wp = W + (long)(tileN + rA) * K   + kO;

    v8f acc[2][2] = {};

    for (int k = 0; k < K; k += 32) {
        v16bf a0 = *(const v16bf*)(Ap + k);
        v16bf a1 = *(const v16bf*)(Ap + (long)16 * lda + k);
        v16bf b0 = *(const v16bf*)(Wp + k);
        v16bf b1 = *(const v16bf*)(Wp + (long)16 * K + k);
        acc[0][0] = __builtin_amdgcn_wmma_f32_16x16x32_bf16(false, a0, false, b0, (short)0, acc[0][0], false, false);
        acc[0][1] = __builtin_amdgcn_wmma_f32_16x16x32_bf16(false, a0, false, b1, (short)0, acc[0][1], false, false);
        acc[1][0] = __builtin_amdgcn_wmma_f32_16x16x32_bf16(false, a1, false, b0, (short)0, acc[1][0], false, false);
        acc[1][1] = __builtin_amdgcn_wmma_f32_16x16x32_bf16(false, a1, false, b1, (short)0, acc[1][1], false, false);
    }

    // C/D fragment layout: VGPR r -> row = r + (lane>=16)*8, col = lane&15
    const int cr = (lane >> 4) * 8;
    const int cc = lane & 15;
#pragma unroll
    for (int i = 0; i < 2; ++i) {
#pragma unroll
        for (int j = 0; j < 2; ++j) {
            const int n = tileN + j * 16 + cc;
            const float bb = bias ? bias[n] : 0.0f;
#pragma unroll
            for (int r = 0; r < 8; ++r) {
                const int m = tileM + i * 16 + cr + r;
                float v = acc[i][j][r] + bb;
                if (EPI == 0) {
                    ((float*)Cout)[(long)m * ldc + n] = v;
                } else {
                    ((bf16*)Cout)[(long)m * ldc + n] = (bf16)(v > 0.0f ? v : 0.0f);
                }
            }
        }
    }
}

// ---------------------------------------------------------------------------
// Big-tile bf16 WMMA GEMM for the large layers (N multiple of 128, M=512).
// Block = 128 threads (4 waves, 2x2). Wave tile = 64x64 (4x4 accumulators =
// 128 acc VGPRs). Block tile = 128(M) x 128(N). Grid = (N/128, M/128).
// Per K=32 step: 8 fragment loads feed 16 v_wmma -> 2x the compute-per-byte
// of the small-tile kernel; keeps the matrix pipe fed from one wave.
// ---------------------------------------------------------------------------
template <int EPI>
__global__ __launch_bounds__(128)
void gemm_wmma_big(const bf16* __restrict__ A, int lda,
                   const bf16* __restrict__ W,
                   const float* __restrict__ bias,
                   void* __restrict__ Cout, int ldc, int K)
{
    const int wave  = threadIdx.x >> 5;
    const int lane  = threadIdx.x & 31;
    const int tileM = blockIdx.y * 128 + (wave >> 1) * 64;
    const int tileN = blockIdx.x * 128 + (wave & 1) * 64;

    const int rA = lane & 15;
    const int kO = (lane >> 4) * 16;

    const bf16* Ap = A + (long)(tileM + rA) * lda + kO;
    const bf16* Wp = W + (long)(tileN + rA) * K   + kO;

    v8f acc[4][4] = {};

    for (int k = 0; k < K; k += 32) {
        v16bf a[4], b[4];
#pragma unroll
        for (int i = 0; i < 4; ++i) {
            a[i] = *(const v16bf*)(Ap + (long)(16 * i) * lda + k);
            b[i] = *(const v16bf*)(Wp + (long)(16 * i) * K   + k);
        }
#pragma unroll
        for (int i = 0; i < 4; ++i) {
#pragma unroll
            for (int j = 0; j < 4; ++j) {
                acc[i][j] = __builtin_amdgcn_wmma_f32_16x16x32_bf16(
                    false, a[i], false, b[j], (short)0, acc[i][j], false, false);
            }
        }
    }

    const int cr = (lane >> 4) * 8;
    const int cc = lane & 15;
#pragma unroll
    for (int i = 0; i < 4; ++i) {
#pragma unroll
        for (int j = 0; j < 4; ++j) {
            const int n = tileN + j * 16 + cc;
            const float bb = bias ? bias[n] : 0.0f;
#pragma unroll
            for (int r = 0; r < 8; ++r) {
                const int m = tileM + i * 16 + cr + r;
                float v = acc[i][j][r] + bb;
                if (EPI == 0) {
                    ((float*)Cout)[(long)m * ldc + n] = v;
                } else {
                    ((bf16*)Cout)[(long)m * ldc + n] = (bf16)(v > 0.0f ? v : 0.0f);
                }
            }
        }
    }
}

// ---------------------------------------------------------------------------
// Elementwise helpers / kernels
// ---------------------------------------------------------------------------
__device__ __forceinline__ float sigm_(float x) { return 1.0f / (1.0f + expf(-x)); }
__device__ __forceinline__ float softplus_(float x) { return x > 20.0f ? x : log1pf(expf(x)); }

__global__ void f32_to_bf16_k(const float* __restrict__ s, bf16* __restrict__ d, int n) {
    int i = blockIdx.x * blockDim.x + threadIdx.x;
    if (i < n) d[i] = (bf16)s[i];
}
__global__ void copy_f32_k(const float* __restrict__ s, float* __restrict__ d, int n) {
    int i = blockIdx.x * blockDim.x + threadIdx.x;
    if (i < n) d[i] = s[i];
}

// init z (bf16) and h (f32 master + bf16) from broadcast z_0 / h_0
__global__ void init_state_k(const float* __restrict__ z0, const float* __restrict__ h0,
                             bf16* __restrict__ zb, float* __restrict__ hf, bf16* __restrict__ hb) {
    int i = blockIdx.x * blockDim.x + threadIdx.x;  // BB*RR threads
    if (i < BB * ZZ) zb[i] = (bf16)z0[i & (ZZ - 1)];
    float h = h0[i & (RR - 1)];
    hf[i] = h;
    hb[i] = (bf16)h;
}

// GRU gating: gi,gh are [BB,3R] f32; updates hf (f32) and hb (bf16)
__global__ void gru_elem_k(const float* __restrict__ GI, const float* __restrict__ GH,
                           float* __restrict__ hf, bf16* __restrict__ hb) {
    int idx = blockIdx.x * blockDim.x + threadIdx.x;  // BB*RR
    int m = idx >> 10, j = idx & (RR - 1);
    const float* gi = GI + (long)m * (3 * RR);
    const float* gh = GH + (long)m * (3 * RR);
    float r = sigm_(gi[j] + gh[j]);
    float z = sigm_(gi[j + RR] + gh[j + RR]);
    float n = tanhf(gi[j + 2 * RR] + r * gh[j + 2 * RR]);
    float h = (1.0f - z) * n + z * hf[idx];
    hf[idx] = h;
    hb[idx] = (bf16)h;
}

// trans combine: z_loc = (1-sig(g))*lin + sig(g)*prop ; prop_relu for scale GEMM
__global__ void trans_elem_k(const float* __restrict__ Tg, const float* __restrict__ Tp,
                             const float* __restrict__ Tl, float* __restrict__ Zloc,
                             bf16* __restrict__ PropRelu) {
    int idx = blockIdx.x * blockDim.x + threadIdx.x;  // BB*ZZ
    float g = sigm_(Tg[idx]);
    float p = Tp[idx];
    Zloc[idx] = (1.0f - g) * Tl[idx] + g * p;
    PropRelu[idx] = (bf16)(p > 0.0f ? p : 0.0f);
}

// reparameterized sample: z_t = z_loc + softplus(Sraw) * eps[:,t,:]
__global__ void sample_elem_k(const float* __restrict__ Zloc, const float* __restrict__ Sraw,
                              const float* __restrict__ eps, int t, bf16* __restrict__ zb) {
    int idx = blockIdx.x * blockDim.x + threadIdx.x;  // BB*ZZ
    int m = idx >> 7, n = idx & (ZZ - 1);
    float z = Zloc[idx] + softplus_(Sraw[idx]) * eps[((long)m * TT + t) * ZZ + n];
    zb[idx] = (bf16)z;
}

// obs combine: x_loc -> out[:,t,0:64]; relu(prop) for scale GEMM
__global__ void obs_elem_k(const float* __restrict__ Og, const float* __restrict__ Op,
                           const float* __restrict__ Ol, float* __restrict__ out, int t,
                           bf16* __restrict__ PropRelu) {
    int idx = blockIdx.x * blockDim.x + threadIdx.x;  // BB*XX
    int m = idx >> 6, n = idx & (XX - 1);
    float g = sigm_(Og[idx]);
    float p = Op[idx];
    out[((long)m * TT + t) * (2 * XX) + n] = (1.0f - g) * Ol[idx] + g * p;
    PropRelu[idx] = (bf16)(p > 0.0f ? p : 0.0f);
}

// x_scale -> out[:,t,64:128]
__global__ void oscale_elem_k(const float* __restrict__ OS, float* __restrict__ out, int t) {
    int idx = blockIdx.x * blockDim.x + threadIdx.x;  // BB*XX
    int m = idx >> 6, n = idx & (XX - 1);
    out[((long)m * TT + t) * (2 * XX) + XX + n] = softplus_(OS[idx]);
}

// ---------------------------------------------------------------------------
// Host launcher
// ---------------------------------------------------------------------------
extern "C" void kernel_launch(void* const* d_in, const int* in_sizes, int n_in,
                              void* d_out, int out_size, void* d_ws, size_t ws_size,
                              hipStream_t stream) {
    (void)in_sizes; (void)n_in; (void)out_size; (void)ws_size;

    const float* eps       = (const float*)d_in[1];
    const float* z0        = (const float*)d_in[2];
    const float* h0        = (const float*)d_in[3];
    const float* gru_b_ih  = (const float*)d_in[5];
    const float* gru_b_hh  = (const float*)d_in[7];
    const float* t_ghz_b   = (const float*)d_in[11];
    const float* t_phz_b   = (const float*)d_in[15];
    const float* t_loc_b   = (const float*)d_in[17];
    const float* t_scale_b = (const float*)d_in[19];
    const float* o_ghz_b   = (const float*)d_in[23];
    const float* o_phz_b   = (const float*)d_in[27];
    const float* o_loc_b   = (const float*)d_in[29];
    const float* o_scale_b = (const float*)d_in[31];
    float* out = (float*)d_out;

    // ---- workspace carve-out (~35 MB) ----
    char* ws = (char*)d_ws;
    size_t off = 0;
    auto take = [&](size_t bytes) -> char* {
        off = (off + 255) & ~(size_t)255;
        char* p = ws + off;
        off += bytes;
        return p;
    };
    // bf16 weight copies ([N,K] row-major, K-contiguous)
    bf16* w_gru_ih = (bf16*)take((size_t)3 * RR * ZZ * 2);       // 3072 x 128
    bf16* w_gru_hh = (bf16*)take((size_t)3 * RR * RR * 2);       // 3072 x 1024
    bf16* w_t_ih   = (bf16*)take((size_t)2 * RR * RR * 2);       // packed 2048 x 1024
    bf16* w_t_ghz  = (bf16*)take((size_t)ZZ * RR * 2);           // 128 x 1024
    bf16* w_t_phz  = (bf16*)take((size_t)ZZ * RR * 2);
    bf16* w_t_loc  = (bf16*)take((size_t)ZZ * RR * 2);
    bf16* w_t_scl  = (bf16*)take((size_t)ZZ * ZZ * 2);           // 128 x 128
    bf16* w_o_ih   = (bf16*)take((size_t)2 * RR * ZZ * 2);       // packed 2048 x 128
    bf16* w_o_ghz  = (bf16*)take((size_t)XX * RR * 2);           // 64 x 1024
    bf16* w_o_phz  = (bf16*)take((size_t)XX * RR * 2);
    bf16* w_o_loc  = (bf16*)take((size_t)XX * ZZ * 2);           // 64 x 128
    bf16* w_o_scl  = (bf16*)take((size_t)XX * XX * 2);           // 64 x 64
    // packed biases for the fused gate/prop hidden GEMMs
    float* b_t_ih = (float*)take((size_t)2 * RR * 4);            // 2048
    float* b_o_ih = (float*)take((size_t)2 * RR * 4);
    // activations / scratch
    float* GI    = (float*)take((size_t)BB * 3 * RR * 4);
    float* GH    = (float*)take((size_t)BB * 3 * RR * 4);
    bf16*  TH    = (bf16*)take((size_t)BB * 2 * RR * 2);         // [512,2048]
    float* Tgate = (float*)take((size_t)BB * ZZ * 4);
    float* Tprop = (float*)take((size_t)BB * ZZ * 4);
    float* Tloc  = (float*)take((size_t)BB * ZZ * 4);
    float* Zloc  = (float*)take((size_t)BB * ZZ * 4);
    float* Sraw  = (float*)take((size_t)BB * ZZ * 4);
    bf16*  TpR   = (bf16*)take((size_t)BB * ZZ * 2);
    bf16*  OH    = (bf16*)take((size_t)BB * 2 * RR * 2);         // [512,2048]
    float* Ogate = (float*)take((size_t)BB * XX * 4);
    float* Oprop = (float*)take((size_t)BB * XX * 4);
    float* Oloc  = (float*)take((size_t)BB * XX * 4);
    float* OSraw = (float*)take((size_t)BB * XX * 4);
    bf16*  OpR   = (bf16*)take((size_t)BB * XX * 2);
    bf16*  zb    = (bf16*)take((size_t)BB * ZZ * 2);
    float* hf    = (float*)take((size_t)BB * RR * 4);
    bf16*  hb    = (bf16*)take((size_t)BB * RR * 2);

    auto conv = [&](const void* s, bf16* d, int n) {
        f32_to_bf16_k<<<(n + 255) / 256, 256, 0, stream>>>((const float*)s, d, n);
    };
    // weight conversions (once per call; weights then live in bf16 in L2)
    conv(d_in[4],  w_gru_ih, 3 * RR * ZZ);
    conv(d_in[6],  w_gru_hh, 3 * RR * RR);
    conv(d_in[8],  w_t_ih,              RR * RR);   // gate_ih
    conv(d_in[12], w_t_ih + RR * RR,    RR * RR);   // prop_ih
    conv(d_in[10], w_t_ghz, ZZ * RR);
    conv(d_in[14], w_t_phz, ZZ * RR);
    conv(d_in[16], w_t_loc, ZZ * RR);
    conv(d_in[18], w_t_scl, ZZ * ZZ);
    conv(d_in[20], w_o_ih,              RR * ZZ);   // o_gate_ih
    conv(d_in[24], w_o_ih + RR * ZZ,    RR * ZZ);   // o_prop_ih
    conv(d_in[22], w_o_ghz, XX * RR);
    conv(d_in[26], w_o_phz, XX * RR);
    conv(d_in[28], w_o_loc, XX * ZZ);
    conv(d_in[30], w_o_scl, XX * XX);
    copy_f32_k<<<(RR + 255) / 256, 256, 0, stream>>>((const float*)d_in[9],  b_t_ih,      RR);
    copy_f32_k<<<(RR + 255) / 256, 256, 0, stream>>>((const float*)d_in[13], b_t_ih + RR, RR);
    copy_f32_k<<<(RR + 255) / 256, 256, 0, stream>>>((const float*)d_in[21], b_o_ih,      RR);
    copy_f32_k<<<(RR + 255) / 256, 256, 0, stream>>>((const float*)d_in[25], b_o_ih + RR, RR);

    init_state_k<<<(BB * RR) / 256, 256, 0, stream>>>(z0, h0, zb, hf, hb);

    const dim3 blk(128);
    auto gs = [](int N) { return dim3(N / 64, BB / 64); };     // small-tile grid
    auto gb = [](int N) { return dim3(N / 128, BB / 128); };   // big-tile grid

    for (int t = 0; t < TT; ++t) {
        // GRU: gi = z @ W_ih^T + b ; gh = h @ W_hh^T + b  (big-tile kernels)
        gemm_wmma_big<0><<<gb(3 * RR), blk, 0, stream>>>(zb, ZZ, w_gru_ih, gru_b_ih, GI, 3 * RR, ZZ);
        gemm_wmma_big<0><<<gb(3 * RR), blk, 0, stream>>>(hb, RR, w_gru_hh, gru_b_hh, GH, 3 * RR, RR);
        gru_elem_k<<<(BB * RR) / 256, 256, 0, stream>>>(GI, GH, hf, hb);

        // trans: fused gate/prop hidden (N=2048, big-tile), then projections
        gemm_wmma_big<1><<<gb(2 * RR), blk, 0, stream>>>(hb, RR, w_t_ih, b_t_ih, TH, 2 * RR, RR);
        gemm_wmma<0><<<gs(ZZ), blk, 0, stream>>>(TH,      2 * RR, w_t_ghz, t_ghz_b, Tgate, ZZ, RR);
        gemm_wmma<0><<<gs(ZZ), blk, 0, stream>>>(TH + RR, 2 * RR, w_t_phz, t_phz_b, Tprop, ZZ, RR);
        gemm_wmma<0><<<gs(ZZ), blk, 0, stream>>>(hb, RR, w_t_loc, t_loc_b, Tloc, ZZ, RR);
        trans_elem_k<<<(BB * ZZ) / 256, 256, 0, stream>>>(Tgate, Tprop, Tloc, Zloc, TpR);
        gemm_wmma<0><<<gs(ZZ), blk, 0, stream>>>(TpR, ZZ, w_t_scl, t_scale_b, Sraw, ZZ, ZZ);
        sample_elem_k<<<(BB * ZZ) / 256, 256, 0, stream>>>(Zloc, Sraw, eps, t, zb);

        // obs: fused gate/prop hidden (N=2048, K=128, big-tile), then projections
        gemm_wmma_big<1><<<gb(2 * RR), blk, 0, stream>>>(zb, ZZ, w_o_ih, b_o_ih, OH, 2 * RR, ZZ);
        gemm_wmma<0><<<gs(XX), blk, 0, stream>>>(OH,      2 * RR, w_o_ghz, o_ghz_b, Ogate, XX, RR);
        gemm_wmma<0><<<gs(XX), blk, 0, stream>>>(OH + RR, 2 * RR, w_o_phz, o_phz_b, Oprop, XX, RR);
        gemm_wmma<0><<<gs(XX), blk, 0, stream>>>(zb, ZZ, w_o_loc, o_loc_b, Oloc, XX, ZZ);
        obs_elem_k<<<(BB * XX) / 256, 256, 0, stream>>>(Ogate, Oprop, Oloc, out, t, OpR);
        gemm_wmma<0><<<gs(XX), blk, 0, stream>>>(OpR, XX, w_o_scl, o_scale_b, OSraw, XX, XX);
        oscale_elem_k<<<(BB * XX) / 256, 256, 0, stream>>>(OSraw, out, t);
    }
}
